// LinearAttention_14577119003080
// MI455X (gfx1250) — matrix operand order repaired and checked by reading, not compiled, not run on hardware
//
#include <hip/hip_runtime.h>
#include <hip/hip_bf16.h>
#include <math.h>

// ---------------------------------------------------------------------------
// LinearAttention for MI455X (gfx1250, wave32, WMMA 16x16x32 f16).
// Flash-attention streaming so the 2304x2304 att matrix never hits HBM.
// K/V tiles staged to LDS via GLOBAL_LOAD_ASYNC_TO_LDS_B128 (ASYNCcnt path);
// builtin signature: (v4i as(1)* gsrc, v4i as(3)* ldst, imm offset, imm cpol).
// ---------------------------------------------------------------------------

typedef __attribute__((ext_vector_type(16))) _Float16 v16h;
typedef __attribute__((ext_vector_type(8)))  float    v8f;

#define EPSV   1e-4f
#define NB     2
#define HEADS  8
#define DHEAD  32
#define SEQ    2304          // 48*48
#define CIN    256
#define CQKV   768

#if __has_builtin(__builtin_amdgcn_global_load_async_to_lds_b128) && \
    __has_builtin(__builtin_amdgcn_s_wait_asynccnt)
#define HAS_ASYNC_LDS 1
typedef int v4i_gcc __attribute__((vector_size(16)));
typedef __attribute__((address_space(1))) v4i_gcc gas_b128;  // global int4
typedef __attribute__((address_space(3))) v4i_gcc las_b128;  // LDS int4
#else
#define HAS_ASYNC_LDS 0
#endif

// workspace layout (bytes, 256-aligned)
static const size_t OFF_WQH = 0;          // 768*256 f16  = 393216
static const size_t OFF_WOH = 393216;     // 256*256 f16  = 131072
static const size_t OFF_Y   = 524288;     // 2*768*2304 f32 = 14155776
static const size_t OFF_Q   = 14680064;   // 2*8*2304*32 f16 = 2359296
static const size_t OFF_K   = 17039360;
static const size_t OFF_V   = 19398656;
static const size_t OFF_YA  = 21757952;   // 2*2304*256 f16 = 2359296 ; total 24117248

// --------------------------- fragment loaders ------------------------------
// A 16x32 f16: lane L holds row M=L%16, K kb..kb+15
// with kb = 16*(L>=16).
__device__ inline v16h load_a16x32(const _Float16* __restrict__ base, int row0, int ld) {
  int lane = threadIdx.x & 31;
  int m = lane & 15, kb = (lane >> 4) * 16;
  const _Float16* p = base + (size_t)(row0 + m) * ld;
  v16h a = {};
#pragma unroll
  for (int j = 0; j < 8; ++j) { a[j] = p[kb + j]; a[8 + j] = p[kb + 8 + j]; }
  return a;
}

// B 32x16 f16: lane L holds column N=L%16, K rows (L>=16)?16..31:0..15.
// Memory layout [N][K] (contiguous along K): value(K=r,N=c) = base[(col0+c)*ld + r]
__device__ inline v16h load_b_nk(const _Float16* __restrict__ base, int col0, int ld) {
  int lane = threadIdx.x & 31;
  int n = lane & 15, kb = (lane >> 4) * 16;
  const _Float16* p = base + (size_t)(col0 + n) * ld + kb;
  v16h b = {};
#pragma unroll
  for (int j = 0; j < 16; ++j) b[j] = p[j];
  return b;
}

// Memory layout [K][N]: value(K=r,N=c) = base[r*ld + col0 + c]
__device__ inline v16h load_b_kn(const _Float16* __restrict__ base, int col0, int ld) {
  int lane = threadIdx.x & 31;
  int n = lane & 15, kb = (lane >> 4) * 16;
  const _Float16* p = base + (size_t)kb * ld + col0 + n;
  v16h b = {};
#pragma unroll
  for (int j = 0; j < 16; ++j) b[j] = p[j * ld];
  return b;
}

// Same but source is fp32 (convert on the fly)
__device__ inline v16h load_b_kn_f32(const float* __restrict__ base, int col0, int ld) {
  int lane = threadIdx.x & 31;
  int n = lane & 15, kb = (lane >> 4) * 16;
  const float* p = base + (size_t)kb * ld + col0 + n;
  v16h b = {};
#pragma unroll
  for (int j = 0; j < 16; ++j) b[j] = (_Float16)p[j * ld];
  return b;
}

__device__ inline v8f wmma_f16(v16h a, v16h b, v8f c) {
  return __builtin_amdgcn_wmma_f32_16x16x32_f16(false, a, false, b, (short)0, c, false, false);
}

// --------------------------- kernel 1: weight norm -------------------------
// wn = w / (eps + ||w||/16) / 16 ; one block per output row, 256 threads.
__global__ void wnorm_kernel(const float* __restrict__ w, _Float16* __restrict__ wh) {
  __shared__ float partial[8];
  int row = blockIdx.x, t = threadIdx.x;
  float v = w[(size_t)row * CIN + t];
  float ss = v * v;
#pragma unroll
  for (int m = 16; m >= 1; m >>= 1) ss += __shfl_xor(ss, m, 32);
  if ((t & 31) == 0) partial[t >> 5] = ss;
  __syncthreads();
  float tot = 0.f;
#pragma unroll
  for (int j = 0; j < 8; ++j) tot += partial[j];
  float scale = 1.0f / ((EPSV + sqrtf(tot * (1.0f / 256.0f))) * 16.0f);
  wh[(size_t)row * CIN + t] = (_Float16)(v * scale);
}

// --------------------------- kernel 2: qkv projection ----------------------
// y[n,o,s] = sum_i wn[o,i] x[n,i,s] ; one wave -> one 16x16 tile, K loop 8x32.
__global__ void qkv_proj_kernel(const _Float16* __restrict__ wq,
                                const float* __restrict__ x,
                                float* __restrict__ y) {
  int wave = (int)((blockIdx.x * blockDim.x + threadIdx.x) >> 5);
  const int MT = CQKV / 16, NT = SEQ / 16;
  int nb = wave / (MT * NT);
  if (nb >= NB) return;
  int r = wave % (MT * NT);
  int mt = r / NT, st = r % NT;
  const float* xb = x + (size_t)nb * CIN * SEQ;
  v8f acc = {};
#pragma unroll
  for (int k0 = 0; k0 < CIN; k0 += 32) {
    v16h a = load_a16x32(wq + k0, mt * 16, CIN);
    v16h b = load_b_kn_f32(xb + (size_t)k0 * SEQ, st * 16, SEQ);
    acc = wmma_f16(a, b, acc);
  }
  int lane = threadIdx.x & 31;
  int n = lane & 15, mb = (lane >> 4) * 8;
  float* yb = y + (size_t)nb * CQKV * SEQ;
#pragma unroll
  for (int i = 0; i < 8; ++i)
    yb[(size_t)(mt * 16 + mb + i) * SEQ + st * 16 + n] = acc[i];
}

// --------------------------- kernel 3: q/k/v normalization -----------------
// channel c = h*96 + d*3 + which; norm over d (32) -> q/k/v [n][h][seq][32] f16
__global__ void qkvnorm_kernel(const float* __restrict__ y,
                               _Float16* __restrict__ q,
                               _Float16* __restrict__ k,
                               _Float16* __restrict__ v) {
  int idx = blockIdx.x * blockDim.x + threadIdx.x;
  if (idx >= NB * HEADS * SEQ) return;
  int s = idx % SEQ;
  int h = (idx / SEQ) % HEADS;
  int nb = idx / (SEQ * HEADS);
  const float* yb = y + (size_t)nb * CQKV * SEQ;
  size_t dbase = (((size_t)(nb * HEADS + h)) * SEQ + s) * DHEAD;
  _Float16* dsts[3] = {q + dbase, k + dbase, v + dbase};
#pragma unroll
  for (int w = 0; w < 3; ++w) {
    float vals[DHEAD]; float ss = 0.f;
#pragma unroll
    for (int d = 0; d < DHEAD; ++d) {
      float vv = yb[(size_t)(h * 96 + d * 3 + w) * SEQ + s];
      vals[d] = vv; ss += vv * vv;
    }
    float sc = 1.0f / (EPSV + sqrtf(ss * (1.0f / 32.0f)));
#pragma unroll
    for (int d = 0; d < DHEAD; ++d) dsts[w][d] = (_Float16)(vals[d] * sc);
  }
}

// --------------------------- kernel 4: flash attention ---------------------
// 1 wave / 16-query strip. Per 32-key chunk: 2 WMMA for S, online softmax via
// half-wave shfl_xor reductions (C/D rows 0-7 live in lanes 0-15), 2 WMMA for PV.
__global__ void __launch_bounds__(32) attn_kernel(const _Float16* __restrict__ q,
                                                  const _Float16* __restrict__ k,
                                                  const _Float16* __restrict__ v,
                                                  _Float16* __restrict__ yatt) {
  __shared__ _Float16 Kt[32 * DHEAD];
  __shared__ _Float16 Vt[32 * DHEAD];
  __shared__ _Float16 Pt[16 * 32];
  int qt = blockIdx.x, h = blockIdx.y, nb = blockIdx.z;
  int lane = threadIdx.x & 31;
  int n = lane & 15, mb = (lane >> 4) * 8;
  size_t headoff = ((size_t)(nb * HEADS + h)) * SEQ * DHEAD;
  const _Float16* kh = k + headoff;
  const _Float16* vh = v + headoff;
  v16h aq = load_a16x32(q + headoff, qt * 16, DHEAD);

  float m[8], l[8];
  v8f acc0 = {}, acc1 = {};
#pragma unroll
  for (int i = 0; i < 8; ++i) { m[i] = -1e30f; l[i] = 0.f; }
  const float sscale = 0.17677669529663687f;  // 1/sqrt(32)

  for (int kc = 0; kc < SEQ / 32; ++kc) {
    int sb = kc * 32;
    const _Float16* gkp = kh + (size_t)(sb + lane) * DHEAD;  // one 64B row / lane
    const _Float16* gvp = vh + (size_t)(sb + lane) * DHEAD;
    _Float16* lkp = Kt + lane * DHEAD;
    _Float16* lvp = Vt + lane * DHEAD;
#if HAS_ASYNC_LDS
    // ASYNCcnt-tracked direct global->LDS copy (no VGPR bounce)
#pragma unroll
    for (int j = 0; j < 4; ++j) {
      __builtin_amdgcn_global_load_async_to_lds_b128(
          (gas_b128*)(gkp + j * 8), (las_b128*)(lkp + j * 8), 0, 0);
      __builtin_amdgcn_global_load_async_to_lds_b128(
          (gas_b128*)(gvp + j * 8), (las_b128*)(lvp + j * 8), 0, 0);
    }
    __builtin_amdgcn_s_wait_asynccnt(0);
#else
    const uint4* gk = (const uint4*)gkp;
    const uint4* gv = (const uint4*)gvp;
    uint4* lk = (uint4*)lkp;
    uint4* lv = (uint4*)lvp;
#pragma unroll
    for (int j = 0; j < 4; ++j) { lk[j] = gk[j]; lv[j] = gv[j]; }
#endif
    // prefetch next chunk's rows while we do softmax + WMMAs on this one
    if (kc + 1 < SEQ / 32) {
      __builtin_prefetch(gkp + 32 * DHEAD, 0, 1);
      __builtin_prefetch(gvp + 32 * DHEAD, 0, 1);
    }
    __syncthreads();

    v16h bk0 = load_b_nk(Kt, 0, DHEAD);
    v16h bk1 = load_b_nk(Kt, 16, DHEAD);
    v8f z = {};
    v8f s0 = wmma_f16(aq, bk0, z);
    v8f s1 = wmma_f16(aq, bk1, z);

    float p0[8], p1[8], alpha[8], rm[8], rs[8];
#pragma unroll
    for (int i = 0; i < 8; ++i) {
      s0[i] *= sscale; s1[i] *= sscale;
      rm[i] = fmaxf(s0[i], s1[i]);
    }
#pragma unroll
    for (int msk = 1; msk < 16; msk <<= 1)
#pragma unroll
      for (int i = 0; i < 8; ++i) rm[i] = fmaxf(rm[i], __shfl_xor(rm[i], msk, 32));
#pragma unroll
    for (int i = 0; i < 8; ++i) {
      float mn = fmaxf(m[i], rm[i]);
      alpha[i] = __expf(m[i] - mn);
      p0[i] = __expf(s0[i] - mn);
      p1[i] = __expf(s1[i] - mn);
      rs[i] = p0[i] + p1[i];
      m[i] = mn;
    }
#pragma unroll
    for (int msk = 1; msk < 16; msk <<= 1)
#pragma unroll
      for (int i = 0; i < 8; ++i) rs[i] += __shfl_xor(rs[i], msk, 32);
#pragma unroll
    for (int i = 0; i < 8; ++i) {
      l[i] = l[i] * alpha[i] + rs[i];
      acc0[i] *= alpha[i];
      acc1[i] *= alpha[i];
      Pt[(mb + i) * 32 + n]      = (_Float16)p0[i];
      Pt[(mb + i) * 32 + 16 + n] = (_Float16)p1[i];
    }
    __syncthreads();

    v16h ap  = load_a16x32(Pt, 0, 32);
    v16h bv0 = load_b_kn(Vt, 0, DHEAD);
    v16h bv1 = load_b_kn(Vt, 16, DHEAD);
    acc0 = wmma_f16(ap, bv0, acc0);
    acc1 = wmma_f16(ap, bv1, acc1);
    __syncthreads();
  }
  // write y_att as [n][seq][256] f16 (B-fragment friendly for out proj)
#pragma unroll
  for (int i = 0; i < 8; ++i) {
    float linv = 1.0f / l[i];
    _Float16* dst = yatt + ((size_t)(nb * SEQ + qt * 16 + mb + i)) * CIN + h * DHEAD;
    dst[n]      = (_Float16)(acc0[i] * linv);
    dst[16 + n] = (_Float16)(acc1[i] * linv);
  }
}

// --------------------------- kernel 5: out proj + residual -----------------
__global__ void outproj_kernel(const _Float16* __restrict__ wo,
                               const _Float16* __restrict__ yatt,
                               const float* __restrict__ x,
                               float* __restrict__ out) {
  int wave = (int)((blockIdx.x * blockDim.x + threadIdx.x) >> 5);
  const int MT = CIN / 16, ST = SEQ / 16;
  int nb = wave / (MT * ST);
  if (nb >= NB) return;
  int r = wave % (MT * ST);
  int mt = r / ST, st = r % ST;
  const _Float16* yb = yatt + (size_t)nb * SEQ * CIN;
  v8f acc = {};
#pragma unroll
  for (int k0 = 0; k0 < CIN; k0 += 32) {
    v16h a = load_a16x32(wo + k0, mt * 16, CIN);
    v16h b = load_b_nk(yb + k0, st * 16, CIN);
    acc = wmma_f16(a, b, acc);
  }
  int lane = threadIdx.x & 31;
  int n = lane & 15, mb = (lane >> 4) * 8;
  const float t = 0.3f, rden = 1.3130643285972254f;  // 1/sqrt(0.58)
#pragma unroll
  for (int i = 0; i < 8; ++i) {
    size_t idx = ((size_t)nb * CIN + mt * 16 + mb + i) * SEQ + st * 16 + n;
    float xv = x[idx];
    out[idx] = (xv + t * (acc[i] - xv)) * rden;
  }
}

// --------------------------- launch ----------------------------------------
extern "C" void kernel_launch(void* const* d_in, const int* in_sizes, int n_in,
                              void* d_out, int out_size, void* d_ws, size_t ws_size,
                              hipStream_t stream) {
  (void)in_sizes; (void)n_in; (void)out_size; (void)ws_size;
  const float* x    = (const float*)d_in[0];
  const float* wqkv = (const float*)d_in[1];
  const float* wout = (const float*)d_in[2];
  float* out = (float*)d_out;
  char* ws = (char*)d_ws;

  _Float16* wqh  = (_Float16*)(ws + OFF_WQH);
  _Float16* woh  = (_Float16*)(ws + OFF_WOH);
  float*    y    = (float*)   (ws + OFF_Y);
  _Float16* q    = (_Float16*)(ws + OFF_Q);
  _Float16* k    = (_Float16*)(ws + OFF_K);
  _Float16* v    = (_Float16*)(ws + OFF_V);
  _Float16* yatt = (_Float16*)(ws + OFF_YA);

  wnorm_kernel<<<CQKV, 256, 0, stream>>>(wqkv, wqh);
  wnorm_kernel<<<CIN, 256, 0, stream>>>(wout, woh);
  // 2 * (768/16) * (2304/16) = 13824 waves, 4 waves/block
  qkv_proj_kernel<<<13824 / 4, 128, 0, stream>>>(wqh, x, y);
  qkvnorm_kernel<<<(NB * HEADS * SEQ) / 256, 256, 0, stream>>>(y, q, k, v);
  attn_kernel<<<dim3(SEQ / 16, HEADS, NB), 32, 0, stream>>>(q, k, v, yatt);
  // 2 * (256/16) * (2304/16) = 4608 waves, 4 waves/block
  outproj_kernel<<<4608 / 4, 128, 0, stream>>>(woh, yatt, x, out);
}
